// TESSLayerGroup_66623532695744
// MI455X (gfx1250) — matrix operand order repaired and checked by reading, not compiled, not run on hardware
//
#include <hip/hip_runtime.h>
#include <hip/hip_bf16.h>
#include <math.h>
#include <stdint.h>

// ---------------------------------------------------------------------------
// Problem constants (match reference)
// ---------------------------------------------------------------------------
constexpr int kB   = 4;
constexpr int kS   = 1024;
constexpr int kD   = 1024;
constexpr int kH   = 16;
constexpr int kHD  = 64;      // kD / kH
constexpr int kL   = 2;
constexpr int kDFF = 4096;
constexpr float kEPS = 1e-12f;

// ---------------------------------------------------------------------------
// Types for CDNA5 WMMA (wave32, 16x16x32 bf16 -> f32)
// ---------------------------------------------------------------------------
typedef __attribute__((ext_vector_type(16))) __bf16 v16bf;
typedef __attribute__((ext_vector_type(8)))  __bf16 v8bf;
typedef __attribute__((ext_vector_type(8)))  float  v8f;

__device__ __forceinline__ __bf16 f2bf(float f) { return (__bf16)f; }

__device__ __forceinline__ v8f wmma_bf16(v16bf a, v16bf b, v8f c) {
  // 8 args: (neg_a, A, neg_b, B, c_mod, C, reuse_a, reuse_b)
  return __builtin_amdgcn_wmma_f32_16x16x32_bf16(false, a, false, b,
                                                 (short)0, c, false, false);
}

__device__ __forceinline__ v16bf cat8(v8bf lo, v8bf hi) {
  return __builtin_shufflevector(lo, hi, 0, 1, 2, 3, 4, 5, 6, 7,
                                 8, 9, 10, 11, 12, 13, 14, 15);
}

// A-matrix 16x32 bf16 fragment from LDS tile stored row-major [m][k].
// Lane (m = lane%16, half = lane/16) needs two contiguous 16B runs:
// k = kofs+half*8+0..7 and k = kofs+16+half*8+0..7  -> 2x ds_load_b128.
__device__ __forceinline__ v16bf frag_a_lds(const __bf16* base, int lda,
                                            int m0, int kofs, int lane) {
  const int m    = m0 + (lane & 15);
  const int half = (lane >> 4) & 1;
  const __bf16* p = base + m * lda + kofs + half * 8;
  v8bf lo = *reinterpret_cast<const v8bf*>(p);
  v8bf hi = *reinterpret_cast<const v8bf*>(p + 16);
  return cat8(lo, hi);
}

// B-matrix 32x16 bf16 fragment from LDS tile stored TRANSPOSED [n][k]:
// lane (n = lane%16, half = lane/16) reads k = kofs+half*8+0..7 and
// k = kofs+16+half*8+0..7 -> 2x b128.
__device__ __forceinline__ v16bf frag_bT_lds(const __bf16* base, int ld,
                                             int n0, int kofs, int lane) {
  const int n  = n0 + (lane & 15);
  const int k0 = kofs + ((lane >> 4) << 3);
  const __bf16* p = base + n * ld + k0;
  v8bf lo = *reinterpret_cast<const v8bf*>(p);
  v8bf hi = *reinterpret_cast<const v8bf*>(p + 16);
  return cat8(lo, hi);
}

// Convert 8 consecutive f32 from global to one b128 LDS store.
__device__ __forceinline__ void stage8(const float* __restrict__ g,
                                       __bf16* __restrict__ l) {
  const float4* s = reinterpret_cast<const float4*>(g);
  float4 x0 = s[0], x1 = s[1];
  v8bf pk;
  pk[0] = f2bf(x0.x); pk[1] = f2bf(x0.y); pk[2] = f2bf(x0.z); pk[3] = f2bf(x0.w);
  pk[4] = f2bf(x1.x); pk[5] = f2bf(x1.y); pk[6] = f2bf(x1.z); pk[7] = f2bf(x1.w);
  *reinterpret_cast<v8bf*>(l) = pk;
}

// Convert 8 consecutive f32 already in LDS to one b128 LDS store.
__device__ __forceinline__ void cvt8_lds(const float* __restrict__ l32,
                                         __bf16* __restrict__ l16) {
  const float4* s = reinterpret_cast<const float4*>(l32);
  float4 x0 = s[0], x1 = s[1];
  v8bf pk;
  pk[0] = f2bf(x0.x); pk[1] = f2bf(x0.y); pk[2] = f2bf(x0.z); pk[3] = f2bf(x0.w);
  pk[4] = f2bf(x1.x); pk[5] = f2bf(x1.y); pk[6] = f2bf(x1.z); pk[7] = f2bf(x1.w);
  *reinterpret_cast<v8bf*>(l16) = pk;
}

// ---------------------------------------------------------------------------
// Tensor Data Mover (TDM): async 2D tile load global(f32) -> LDS.
// Descriptor per CDNA5 ISA. TENSORcnt tracked.
// ---------------------------------------------------------------------------
#if __has_builtin(__builtin_amdgcn_tensor_load_to_lds) && \
    __has_builtin(__builtin_amdgcn_s_wait_tensorcnt)
#define USE_TDM 1
typedef __attribute__((ext_vector_type(4))) unsigned int v4u;
typedef __attribute__((ext_vector_type(8))) int         v8i_;
typedef __attribute__((ext_vector_type(4))) int         v4i_;

// Load a 2D tile [tile_h rows x tile_w f32] from a row-major f32 tensor with
// row stride `row_stride` (elements) into packed LDS at byte offset lds_addr.
__device__ __forceinline__ void tdm_load_tile_f32(unsigned lds_addr,
                                                  const float* gaddr,
                                                  unsigned tile_w,
                                                  unsigned tile_h,
                                                  unsigned row_stride,
                                                  unsigned tensor_rows) {
  const unsigned long long ga = (unsigned long long)(uintptr_t)gaddr;
  v4u g0;
  g0[0] = 1u;                                  // count=1 (valid), user mode
  g0[1] = lds_addr;                            // LDS byte address
  g0[2] = (unsigned)(ga & 0xFFFFFFFFu);        // global_addr[31:0]
  g0[3] = (unsigned)((ga >> 32) & 0x01FFFFFFu) // global_addr[56:32]
          | (2u << 30);                        // type=2 ("image")
  v8i_ g1;
  g1[0] = (int)(2u << 16);                     // wg_mask=0, data_size=2 (4B)
  g1[1] = (int)((row_stride & 0xFFFFu) << 16);            // tensor_dim0[15:0]
  g1[2] = (int)(((row_stride >> 16) & 0xFFFFu)            // tensor_dim0[31:16]
                | ((tensor_rows & 0xFFFFu) << 16));       // tensor_dim1[15:0]
  g1[3] = (int)(((tensor_rows >> 16) & 0xFFFFu)           // tensor_dim1[31:16]
                | ((tile_w & 0xFFFFu) << 16));            // tile_dim0
  g1[4] = (int)(tile_h & 0xFFFFu);                        // tile_dim1, dim2=0
  g1[5] = (int)row_stride;                                // dim0_stride[31:0]
  g1[6] = 0;                                              // dim0_stride[47:32], dim1_stride[15:0]
  g1[7] = 0;                                              // dim1_stride[47:16]
  const v4i_ gz = {};
#if __clang_major__ >= 23
  const v8i_ gz8 = {};
  __builtin_amdgcn_tensor_load_to_lds(g0, g1, gz, gz, gz8, 0);
#else
  __builtin_amdgcn_tensor_load_to_lds(g0, g1, gz, gz, 0);
#endif
}
#else
#define USE_TDM 0
#endif

// ---------------------------------------------------------------------------
// GEMM: C = act(A[MxK] @ W[KxN] + bias[N] (+ Res[MxN]))
// block = 256 threads = 8 waves; block tile 128x128; wave tile 32x64;
// K-step 64 (two WMMA depths per stage). ACT: 0=none, 1=exact GELU.
// ---------------------------------------------------------------------------
template <int ACT, int RES>
__global__ __launch_bounds__(256)
void gemm_wmma_kernel(const float* __restrict__ A,
                      const float* __restrict__ W,
                      const float* __restrict__ bias,
                      const float* __restrict__ Res,
                      float* __restrict__ C,
                      int M, int N, int K) {
  __shared__ __align__(16) __bf16 lA[128][80];   // [m][k], 160B rows
  __shared__ __align__(16) __bf16 lBt[128][80];  // [n][k] transposed

  const int tid  = threadIdx.x;
  const int lane = tid & 31;
  const int w    = tid >> 5;
  const int wm   = w & 3;    // 4 waves along M
  const int wn   = w >> 2;   // 2 waves along N
  const int bm   = blockIdx.y * 128;
  const int bn   = blockIdx.x * 128;

  v8f acc[2][4];
  const v8f vz = {};
#pragma unroll
  for (int i = 0; i < 2; ++i)
#pragma unroll
    for (int j = 0; j < 4; ++j) acc[i][j] = vz;

  const int arow = tid >> 1;          // 0..127
  const int acol = (tid & 1) * 32;    // 0 or 32 within K-step
  const int brow = tid >> 2;          // 0..63  (k within tile)
  const int bcol = (tid & 3) * 32;    // 0..96  (n within tile)

  for (int kt = 0; kt < K; kt += 64) {
    if (kt + 64 < K) {
      __builtin_prefetch(A + (size_t)(bm + arow) * K + (kt + 64) + acol, 0, 0);
      __builtin_prefetch(W + (size_t)(kt + 64 + brow) * N + bn + bcol, 0, 0);
    }
    // Stage A tile (128x64) f32 -> bf16 row-major, packed b128 stores
    {
      const float* src = A + (size_t)(bm + arow) * K + kt + acol;
#pragma unroll
      for (int q = 0; q < 4; ++q)
        stage8(src + q * 8, &lA[arow][acol + q * 8]);
    }
    // Stage W tile (64x128) f32 -> bf16, transposed into [n][k]
    {
      const float4* src =
          reinterpret_cast<const float4*>(W + (size_t)(kt + brow) * N + bn + bcol);
#pragma unroll
      for (int q = 0; q < 8; ++q) {
        float4 x = src[q];
        lBt[bcol + q * 4 + 0][brow] = f2bf(x.x);
        lBt[bcol + q * 4 + 1][brow] = f2bf(x.y);
        lBt[bcol + q * 4 + 2][brow] = f2bf(x.z);
        lBt[bcol + q * 4 + 3][brow] = f2bf(x.w);
      }
    }
    __syncthreads();

#pragma unroll
    for (int kc = 0; kc < 2; ++kc) {
      v16bf af[2], bf[4];
#pragma unroll
      for (int mt = 0; mt < 2; ++mt)
        af[mt] = frag_a_lds(&lA[0][0], 80, wm * 32 + mt * 16, kc * 32, lane);
#pragma unroll
      for (int nt = 0; nt < 4; ++nt)
        bf[nt] = frag_bT_lds(&lBt[0][0], 80, wn * 64 + nt * 16, kc * 32, lane);
#pragma unroll
      for (int mt = 0; mt < 2; ++mt)
#pragma unroll
        for (int nt = 0; nt < 4; ++nt)
          acc[mt][nt] = wmma_bf16(af[mt], bf[nt], acc[mt][nt]);
    }
    __syncthreads();
  }

  // Epilogue: bias (+residual) (+GELU). C layout: lane=N%16, row=vgpr+8*half.
  const int half = lane >> 4;
#pragma unroll
  for (int mt = 0; mt < 2; ++mt) {
#pragma unroll
    for (int nt = 0; nt < 4; ++nt) {
      const int col = bn + wn * 64 + nt * 16 + (lane & 15);
      const float bv = bias[col];
#pragma unroll
      for (int i = 0; i < 8; ++i) {
        const int row = bm + wm * 32 + mt * 16 + half * 8 + i;
        float x = acc[mt][nt][i] + bv;
        if (RES) x += Res[(size_t)row * N + col];
        if (ACT == 1) x = 0.5f * x * (1.0f + erff(x * 0.70710678118654752f));
        C[(size_t)row * N + col] = x;
      }
    }
  }
}

// ---------------------------------------------------------------------------
// Flash attention: one wave (32 threads) per (b, h, 16-query tile).
// Q/K/V are [B*S, D] with head h in columns h*64..h*64+63 (HD=64).
// scores = Q K^T * scale + mask; probs = softmax * head_mask; ctx = probs V.
// K/V tiles are staged by the Tensor Data Mover (raw f32 -> LDS), then
// converted to bf16 WMMA-friendly tiles.
// ---------------------------------------------------------------------------
__global__ __launch_bounds__(32)
void attn_flash_kernel(const float* __restrict__ Q,
                       const float* __restrict__ Km,
                       const float* __restrict__ V,
                       const float* __restrict__ amask,   // [B, S]
                       const float* __restrict__ hmask,   // [H] (layer slice)
                       float* __restrict__ O) {
  const int lane = threadIdx.x;
  const int b  = blockIdx.z;
  const int h  = blockIdx.y;
  const int q0 = blockIdx.x * 16;
  const float scale = 0.125f;          // 1/sqrt(64)
  const float hm = hmask[h];
  const int half = lane >> 4;

  __shared__ __align__(16) __bf16 sQ[16][80];   // [q][dim]
  __shared__ __align__(16) __bf16 sK[32][80];   // [key][dim]  (B^T for scores)
  __shared__ __align__(16) __bf16 sVt[64][40];  // [dim][key]  (B^T for ctx)
  __shared__ __align__(16) __bf16 sP[16][40];   // probs, A-layout staging
#if USE_TDM
  __shared__ __align__(16) float rawK[32][64];  // TDM destination (packed)
  __shared__ __align__(16) float rawV[32][64];
#endif

  // Stage Q tile (16x64): 128 chunks of 8, 4 per lane
#pragma unroll
  for (int q = 0; q < 4; ++q) {
    const int chunk = lane + 32 * q;
    const int r = chunk >> 3, c = (chunk & 7) * 8;
    stage8(Q + (size_t)(b * kS + q0 + r) * kD + h * 64 + c, &sQ[r][c]);
  }
  __syncthreads();
  v16bf qf[2];
#pragma unroll
  for (int kc = 0; kc < 2; ++kc)
    qf[kc] = frag_a_lds(&sQ[0][0], 80, 0, kc * 32, lane);

  v8f ctx[4];
  const v8f vz = {};
#pragma unroll
  for (int t = 0; t < 4; ++t) ctx[t] = vz;
  float rm[8], rl[8];
#pragma unroll
  for (int i = 0; i < 8; ++i) { rm[i] = -3.0e38f; rl[i] = 0.f; }

  for (int kt = 0; kt < kS / 32; ++kt) {
    __syncthreads();
#if USE_TDM
    // TDM: async DMA the 32x64 f32 K and V tiles into packed LDS.
    tdm_load_tile_f32((unsigned)(uintptr_t)&rawK[0][0],
                      Km + (size_t)(b * kS + kt * 32) * kD + h * 64,
                      64u, 32u, (unsigned)kD, (unsigned)(kB * kS));
    tdm_load_tile_f32((unsigned)(uintptr_t)&rawV[0][0],
                      V + (size_t)(b * kS + kt * 32) * kD + h * 64,
                      64u, 32u, (unsigned)kD, (unsigned)(kB * kS));
    __builtin_amdgcn_s_wait_tensorcnt(0);
    __syncthreads();
    // Convert raw f32 tiles -> bf16 WMMA layouts (sK row-major, sVt transposed)
#pragma unroll
    for (int q = 0; q < 8; ++q) {
      const int chunk = lane + 32 * q;           // 256 chunks of 8
      const int r = chunk >> 3, c = (chunk & 7) * 8;
      cvt8_lds(&rawK[r][c], &sK[r][c]);
      const float4* sv = reinterpret_cast<const float4*>(&rawV[r][c]);
      float4 x0 = sv[0], x1 = sv[1];
      sVt[c + 0][r] = f2bf(x0.x); sVt[c + 1][r] = f2bf(x0.y);
      sVt[c + 2][r] = f2bf(x0.z); sVt[c + 3][r] = f2bf(x0.w);
      sVt[c + 4][r] = f2bf(x1.x); sVt[c + 5][r] = f2bf(x1.y);
      sVt[c + 6][r] = f2bf(x1.z); sVt[c + 7][r] = f2bf(x1.w);
    }
#else
    // Fallback: direct global -> LDS staging with conversion
#pragma unroll
    for (int q = 0; q < 8; ++q) {
      const int chunk = lane + 32 * q;
      const int r = chunk >> 3, c = (chunk & 7) * 8;
      stage8(Km + (size_t)(b * kS + kt * 32 + r) * kD + h * 64 + c, &sK[r][c]);
      const float4* sv = reinterpret_cast<const float4*>(
          V + (size_t)(b * kS + kt * 32 + r) * kD + h * 64 + c);
      float4 x0 = sv[0], x1 = sv[1];
      sVt[c + 0][r] = f2bf(x0.x); sVt[c + 1][r] = f2bf(x0.y);
      sVt[c + 2][r] = f2bf(x0.z); sVt[c + 3][r] = f2bf(x0.w);
      sVt[c + 4][r] = f2bf(x1.x); sVt[c + 5][r] = f2bf(x1.y);
      sVt[c + 6][r] = f2bf(x1.z); sVt[c + 7][r] = f2bf(x1.w);
    }
#endif
    __syncthreads();

    // scores(16q x 32keys): A=Q(16x64), B=K^T -> sK stored [key][dim]=[n][k]
    v8f sc[2];
#pragma unroll
    for (int nt = 0; nt < 2; ++nt) {
      v8f s = vz;
#pragma unroll
      for (int kc = 0; kc < 2; ++kc) {
        v16bf bf = frag_bT_lds(&sK[0][0], 80, nt * 16, kc * 32, lane);
        s = wmma_bf16(qf[kc], bf, s);
      }
      const float msk = amask[(size_t)b * kS + kt * 32 + nt * 16 + (lane & 15)];
#pragma unroll
      for (int i = 0; i < 8; ++i) s[i] = s[i] * scale + msk;
      sc[nt] = s;
    }

    // Online softmax (row stats live in the 16 lanes of each half)
    float p0[8], p1[8];
#pragma unroll
    for (int i = 0; i < 8; ++i) {
      float t = fmaxf(sc[0][i], sc[1][i]);
#pragma unroll
      for (int off = 1; off < 16; off <<= 1) t = fmaxf(t, __shfl_xor(t, off, 32));
      const float nm  = fmaxf(rm[i], t);
      const float fac = __expf(rm[i] - nm);
      rm[i] = nm;
      const float a  = __expf(sc[0][i] - nm);
      const float c2 = __expf(sc[1][i] - nm);
      float ps = a + c2;
#pragma unroll
      for (int off = 1; off < 16; off <<= 1) ps += __shfl_xor(ps, off, 32);
      rl[i] = rl[i] * fac + ps;
      p0[i] = a; p1[i] = c2;
#pragma unroll
      for (int t4 = 0; t4 < 4; ++t4) ctx[t4][i] *= fac;
    }

    // C-layout -> A-layout via LDS (head_mask applied post-softmax)
#pragma unroll
    for (int i = 0; i < 8; ++i) {
      const int m = i + half * 8;
      sP[m][(lane & 15)]      = f2bf(p0[i] * hm);
      sP[m][16 + (lane & 15)] = f2bf(p1[i] * hm);
    }
    __syncthreads();

    const v16bf pf = frag_a_lds(&sP[0][0], 40, 0, 0, lane);
#pragma unroll
    for (int t4 = 0; t4 < 4; ++t4) {
      v16bf vf = frag_bT_lds(&sVt[0][0], 40, t4 * 16, 0, lane);
      ctx[t4] = wmma_bf16(pf, vf, ctx[t4]);
    }
  }

  // Normalize and write ctx -> O[b, s, h*64 + d]
#pragma unroll
  for (int t4 = 0; t4 < 4; ++t4) {
#pragma unroll
    for (int i = 0; i < 8; ++i) {
      const int srow = q0 + i + half * 8;
      const int col  = h * 64 + t4 * 16 + (lane & 15);
      O[(size_t)(b * kS + srow) * kD + col] = ctx[t4][i] / rl[i];
    }
  }
}

// ---------------------------------------------------------------------------
// LayerNorm over last dim (D=1024), one 256-thread block per row.
// ---------------------------------------------------------------------------
__global__ __launch_bounds__(256)
void layernorm_kernel(const float* __restrict__ x,
                      const float* __restrict__ g,
                      const float* __restrict__ bta,
                      float* __restrict__ out, int Dn) {
  const int row = blockIdx.x;
  const int tid = threadIdx.x;
  const float* xr = x + (size_t)row * Dn;

  float s = 0.f, s2 = 0.f;
  for (int i = tid; i < Dn; i += 256) {
    const float v = xr[i];
    s += v; s2 += v * v;
  }
#pragma unroll
  for (int off = 16; off; off >>= 1) {
    s  += __shfl_xor(s, off, 32);
    s2 += __shfl_xor(s2, off, 32);
  }
  __shared__ float shs[8], shs2[8];
  if ((tid & 31) == 0) { shs[tid >> 5] = s; shs2[tid >> 5] = s2; }
  __syncthreads();
  if (tid < 32) {
    float a  = (tid < 8) ? shs[tid]  : 0.f;
    float a2 = (tid < 8) ? shs2[tid] : 0.f;
#pragma unroll
    for (int off = 4; off; off >>= 1) {
      a  += __shfl_xor(a, off, 32);
      a2 += __shfl_xor(a2, off, 32);
    }
    if (tid == 0) { shs[0] = a; shs2[0] = a2; }
  }
  __syncthreads();
  const float mean = shs[0] / (float)Dn;
  const float var  = shs2[0] / (float)Dn - mean * mean;
  const float rstd = rsqrtf(var + kEPS);
  for (int i = tid; i < Dn; i += 256) {
    out[(size_t)row * Dn + i] = (xr[i] - mean) * rstd * g[i] + bta[i];
  }
}

// ---------------------------------------------------------------------------
// Host-side sequencing
// ---------------------------------------------------------------------------
extern "C" void kernel_launch(void* const* d_in, const int* in_sizes, int n_in,
                              void* d_out, int out_size, void* d_ws, size_t ws_size,
                              hipStream_t stream) {
  const float* hs_in   = (const float*)d_in[0];
  const float* amask   = (const float*)d_in[1];
  const float* hmask   = (const float*)d_in[2];
  const float* q_w     = (const float*)d_in[3];
  const float* q_b     = (const float*)d_in[4];
  const float* k_w     = (const float*)d_in[5];
  const float* k_b     = (const float*)d_in[6];
  const float* v_w     = (const float*)d_in[7];
  const float* v_b     = (const float*)d_in[8];
  const float* o_w     = (const float*)d_in[9];
  const float* o_b     = (const float*)d_in[10];
  const float* aln_g   = (const float*)d_in[11];
  const float* aln_b   = (const float*)d_in[12];
  const float* w1      = (const float*)d_in[13];
  const float* b1      = (const float*)d_in[14];
  const float* w2      = (const float*)d_in[15];
  const float* b2      = (const float*)d_in[16];
  const float* fln_g   = (const float*)d_in[17];
  const float* fln_b   = (const float*)d_in[18];

  const size_t M = (size_t)kB * kS;   // 4096 rows
  float* ws = (float*)d_ws;
  float* Qb      = ws;                       // M*D
  float* Kb      = Qb + M * kD;              // M*D   (also FFN-sum tmp)
  float* Vb      = Kb + M * kD;              // M*D
  float* Ctx     = Vb + M * kD;              // M*D
  float* AttnOut = Ctx + M * kD;             // M*D
  float* Ff      = AttnOut + M * kD;         // M*DFF
  float* HsBuf   = Ff + M * kDFF;            // M*D

  const dim3 blk(256);
  const dim3 gD (kD  / 128, (int)(M / 128));  // N=1024 gemms
  const dim3 gFF(kDFF / 128, (int)(M / 128)); // N=4096 gemm
  const dim3 gAttn(kS / 16, kH, kB);
  const dim3 gLN((unsigned)M);

  for (int i = 0; i < kL; ++i) {
    const float* hs  = (i == 0) ? hs_in : HsBuf;
    float* hs_next   = (i == kL - 1) ? (float*)d_out : HsBuf;

    // Q, K, V projections
    gemm_wmma_kernel<0, 0><<<gD, blk, 0, stream>>>(hs, q_w, q_b, nullptr, Qb,
                                                   (int)M, kD, kD);
    gemm_wmma_kernel<0, 0><<<gD, blk, 0, stream>>>(hs, k_w, k_b, nullptr, Kb,
                                                   (int)M, kD, kD);
    gemm_wmma_kernel<0, 0><<<gD, blk, 0, stream>>>(hs, v_w, v_b, nullptr, Vb,
                                                   (int)M, kD, kD);
    // Attention (flash, per 16-query tile; K/V tiles via TDM)
    attn_flash_kernel<<<gAttn, dim3(32), 0, stream>>>(Qb, Kb, Vb, amask,
                                                      hmask + (size_t)i * kH, Ctx);
    // attn_sum = hs + ctx @ o_w + o_b   (into Qb, then LN -> AttnOut)
    gemm_wmma_kernel<0, 1><<<gD, blk, 0, stream>>>(Ctx, o_w, o_b, hs, Qb,
                                                   (int)M, kD, kD);
    layernorm_kernel<<<gLN, blk, 0, stream>>>(Qb, aln_g, aln_b, AttnOut, kD);
    // FFN: h = gelu(attn_out @ w1 + b1)
    gemm_wmma_kernel<1, 0><<<gFF, blk, 0, stream>>>(
        AttnOut, w1 + (size_t)i * kD * kDFF, b1 + (size_t)i * kDFF, nullptr, Ff,
        (int)M, kDFF, kD);
    // ffn_sum = attn_out + h @ w2 + b2  (into Kb, then LN -> hs_next)
    gemm_wmma_kernel<0, 1><<<gD, blk, 0, stream>>>(
        Ff, w2 + (size_t)i * kDFF * kD, b2 + (size_t)i * kD, AttnOut, Kb,
        (int)M, kD, kDFF);
    layernorm_kernel<<<gLN, blk, 0, stream>>>(Kb, fln_g + (size_t)i * kD,
                                              fln_b + (size_t)i * kD, hs_next, kD);
  }
}
